// BahdanauAttention_24515673325998
// MI455X (gfx1250) — matrix operand order repaired and checked by reading, not compiled, not run on hardware
//
#include <hip/hip_runtime.h>
#include <hip/hip_bf16.h>

typedef __attribute__((ext_vector_type(16))) __bf16 v16bf;
typedef __attribute__((ext_vector_type(8)))  float  v8f;
typedef unsigned short u16;
typedef unsigned int   u32;

#define B_  32
#define S_  2048
#define E_  1024
#define D_  1024
#define NEG_INF_ (-1e9f)

// ---------- f32 -> bf16 (RNE) streaming converter ----------

__device__ __forceinline__ u32 pk2bf(float lo, float hi) {
  union { float f; u32 u; } a, b; a.f = lo; b.f = hi;
  u32 ua = a.u + (0x7FFFu + ((a.u >> 16) & 1u));
  u32 ub = b.u + (0x7FFFu + ((b.u >> 16) & 1u));
  return (ua >> 16) | (ub & 0xFFFF0000u);
}

// n8 = n/8 ; converts 8 floats per thread
__global__ void __launch_bounds__(256) f32_to_bf16_kernel(
    const float* __restrict__ src, u32* __restrict__ dst, int n8)
{
  const int i = blockIdx.x * 256 + threadIdx.x;
  if (i >= n8) return;
  const float4* s = (const float4*)src + (size_t)i * 2;
  float4 x0 = s[0], x1 = s[1];
  uint4 o;
  o.x = pk2bf(x0.x, x0.y);
  o.y = pk2bf(x0.z, x0.w);
  o.z = pk2bf(x1.x, x1.y);
  o.w = pk2bf(x1.z, x1.w);
  ((uint4*)dst)[i] = o;
}

// ---------- bf16 fragment loaders (pure b128 loads, no VALU) ----------

union frag_u { uint4 q[2]; v16bf v; };

// A fragment (16x32 MxK) from row-major bf16 [*, ldk]:
// lanes 0-15: row M=lane, K [k..k+7],[k+16..k+23]; lanes 16-31: row lane-16, K +8.
__device__ __forceinline__ v16bf load_afrag_bf(const u16* __restrict__ src, int ldk,
                                               int mbase, int k, int lane) {
  const int l15 = lane & 15;
  const int ka  = k + ((lane < 16) ? 0 : 8);
  const u16* p  = src + (size_t)(mbase + l15) * ldk + ka;
  frag_u f;
  f.q[0] = *(const uint4*)(p);        // K [ka .. ka+7]
  f.q[1] = *(const uint4*)(p + 16);   // K [ka+16 .. ka+23]
  return f.v;
}

// B fragment (32x16 KxN) for NT GEMM from row-major bf16 weights [N, ldk]:
// lanes 0-15: col N=lane, K [k..k+7],[k+16..k+23]; lanes 16-31: col lane-16, K +8.
__device__ __forceinline__ v16bf load_bfrag_bf(const u16* __restrict__ w, int ldk,
                                               int nbase, int k, int lane) {
  const int l15 = lane & 15;
  const int kr  = k + ((lane < 16) ? 0 : 8);
  const u16* p  = w + (size_t)(nbase + l15) * ldk + kr;
  frag_u f;
  f.q[0] = *(const uint4*)(p);        // K [kr .. kr+7]
  f.q[1] = *(const uint4*)(p + 16);   // K [kr+16 .. kr+23]
  return f.v;
}

// ---------- kernel 1: dec[b,d] = decoder_hidden @ W_s^T (bf16 operands) ----------

__global__ void __launch_bounds__(32) dec_proj_kernel(
    const u16* __restrict__ dhb,   // [B, D] bf16
    const u16* __restrict__ wsb,   // [D, D] bf16
    float* __restrict__ dec)       // [B, D] f32
{
  const int lane  = threadIdx.x & 31;
  const int mbase = (blockIdx.x >> 6) * 16;
  const int nbase = (blockIdx.x & 63) * 16;
  v8f acc = {};
  for (int k = 0; k < D_; k += 32) {
    v16bf a = load_afrag_bf(dhb, D_, mbase, k, lane);
    v16bf b = load_bfrag_bf(wsb, D_, nbase, k, lane);
    acc = __builtin_amdgcn_wmma_f32_16x16x32_bf16(false, a, false, b,
                                                  (short)0, acc, false, false);
  }
  const int col   = nbase + (lane & 15);
  const int rbase = mbase + ((lane < 16) ? 0 : 8);
#pragma unroll
  for (int r = 0; r < 8; ++r)
    dec[(size_t)(rbase + r) * D_ + col] = acc[r];
}

// ---------- kernel 2: fused energy GEMM + tanh + v-dot ----------
// Workgroup (8 waves) covers M=64 rows of flattened [B*S]; wave w owns
// D-columns [w*128, w*128+128) for all four 16-row M-tiles (B-fragment reuse x4).

__global__ void __launch_bounds__(256) energy_kernel(
    const u16* __restrict__ encb,   // [B*S, E] bf16
    const u16* __restrict__ whb,    // [D, E] bf16
    const float* __restrict__ dec,  // [B, D]
    const float* __restrict__ v,    // [D]
    float* __restrict__ energy)     // [B, S]
{
  const int lane  = threadIdx.x & 31;
  const int wave  = threadIdx.x >> 5;      // 0..7
  const int mbase = blockIdx.x * 64;       // flat row base (S%64==0 -> single b)
  const int b     = mbase >> 11;
  const int sbase = mbase & (S_ - 1);
  const int l15   = lane & 15;
  const int nwave = wave * 128;

  v8f acc[4][8];
#pragma unroll
  for (int g = 0; g < 4; ++g)
#pragma unroll
    for (int t = 0; t < 8; ++t) acc[g][t] = (v8f){};

  for (int k = 0; k < E_; k += 32) {
    v16bf a0 = load_afrag_bf(encb, E_, mbase,      k, lane);
    v16bf a1 = load_afrag_bf(encb, E_, mbase + 16, k, lane);
    v16bf a2 = load_afrag_bf(encb, E_, mbase + 32, k, lane);
    v16bf a3 = load_afrag_bf(encb, E_, mbase + 48, k, lane);
#pragma unroll
    for (int t = 0; t < 8; ++t) {
      v16bf bf = load_bfrag_bf(whb, E_, nwave + t * 16, k, lane);
      acc[0][t] = __builtin_amdgcn_wmma_f32_16x16x32_bf16(false, a0, false, bf,
                                                          (short)0, acc[0][t], false, false);
      acc[1][t] = __builtin_amdgcn_wmma_f32_16x16x32_bf16(false, a1, false, bf,
                                                          (short)0, acc[1][t], false, false);
      acc[2][t] = __builtin_amdgcn_wmma_f32_16x16x32_bf16(false, a2, false, bf,
                                                          (short)0, acc[2][t], false, false);
      acc[3][t] = __builtin_amdgcn_wmma_f32_16x16x32_bf16(false, a3, false, bf,
                                                          (short)0, acc[3][t], false, false);
    }
  }

  __shared__ float red[8][64];

#pragma unroll
  for (int g = 0; g < 4; ++g) {
    float eacc[8];
#pragma unroll
    for (int r = 0; r < 8; ++r) eacc[r] = 0.f;
#pragma unroll
    for (int t = 0; t < 8; ++t) {
      const int n    = nwave + t * 16 + l15;
      const float dv = dec[b * D_ + n];
      const float vv = v[n];
#pragma unroll
      for (int r = 0; r < 8; ++r)
        eacc[r] += tanhf(acc[g][t][r] + dv) * vv;
    }
    // reduce across the 16 column-lanes (halves independent)
#pragma unroll
    for (int r = 0; r < 8; ++r) {
      float x = eacc[r];
      x += __shfl_xor(x, 1);
      x += __shfl_xor(x, 2);
      x += __shfl_xor(x, 4);
      x += __shfl_xor(x, 8);
      eacc[r] = x;
    }
    if (lane == 0) {
#pragma unroll
      for (int r = 0; r < 8; ++r) red[wave][g * 16 + r] = eacc[r];
    } else if (lane == 16) {
#pragma unroll
      for (int r = 0; r < 8; ++r) red[wave][g * 16 + 8 + r] = eacc[r];
    }
  }
  __syncthreads();
  if (threadIdx.x < 64) {
    float s = 0.f;
#pragma unroll
    for (int w = 0; w < 8; ++w) s += red[w][threadIdx.x];
    energy[b * S_ + sbase + threadIdx.x] = s;
  }
}

// ---------- kernel 3: masked softmax over S (in place) ----------

__global__ void __launch_bounds__(256) softmax_kernel(
    float* __restrict__ attn,        // [B, S] in: energy, out: softmax
    const int* __restrict__ mask)    // [B, S]
{
  __shared__ float sh[8];
  const int b = blockIdx.x;
  const int t = threadIdx.x;

  float e[8];
  float m = -3.4e38f;
#pragma unroll
  for (int i = 0; i < 8; ++i) {
    const int s = t + i * 256;
    float x = (mask[b * S_ + s] == 0) ? NEG_INF_ : attn[b * S_ + s];
    e[i] = x;
    m = fmaxf(m, x);
  }
#pragma unroll
  for (int off = 16; off > 0; off >>= 1) m = fmaxf(m, __shfl_xor(m, off));
  if ((t & 31) == 0) sh[t >> 5] = m;
  __syncthreads();
  float mm = sh[0];
#pragma unroll
  for (int w = 1; w < 8; ++w) mm = fmaxf(mm, sh[w]);
  __syncthreads();

  float sum = 0.f;
#pragma unroll
  for (int i = 0; i < 8; ++i) { e[i] = __expf(e[i] - mm); sum += e[i]; }
#pragma unroll
  for (int off = 16; off > 0; off >>= 1) sum += __shfl_xor(sum, off);
  if ((t & 31) == 0) sh[t >> 5] = sum;
  __syncthreads();
  float tot = 0.f;
#pragma unroll
  for (int w = 0; w < 8; ++w) tot += sh[w];

  const float inv = 1.0f / tot;
#pragma unroll
  for (int i = 0; i < 8; ++i) attn[b * S_ + t + i * 256] = e[i] * inv;
}

// ---------- kernel 4: context[b,e] = sum_s attn[b,s] * enc[b,s,e] (f32) ----------

__global__ void __launch_bounds__(256) context_kernel(
    const float* __restrict__ enc,   // [B, S, E]
    const float* __restrict__ attn,  // [B, S]
    float* __restrict__ ctx)         // [B, E]
{
  const int b  = blockIdx.x;
  const int e0 = blockIdx.y * 256 + threadIdx.x;
  const float* base = enc + (size_t)b * S_ * E_ + e0;
  const float* ab   = attn + b * S_;
  float acc = 0.f;
  for (int s = 0; s < S_; s += 4) {
    __builtin_prefetch(base + (size_t)(s + 16) * E_, 0, 1);
    float4 a = *(const float4*)(ab + s);
    acc += a.x * base[(size_t)(s + 0) * E_];
    acc += a.y * base[(size_t)(s + 1) * E_];
    acc += a.z * base[(size_t)(s + 2) * E_];
    acc += a.w * base[(size_t)(s + 3) * E_];
  }
  ctx[b * E_ + e0] = acc;
}

// ---------- launch ----------

extern "C" void kernel_launch(void* const* d_in, const int* in_sizes, int n_in,
                              void* d_out, int out_size, void* d_ws, size_t ws_size,
                              hipStream_t stream) {
  const float* dh   = (const float*)d_in[0];  // [B, D]
  const float* enc  = (const float*)d_in[1];  // [B, S, E]
  const int*   mask = (const int*)d_in[2];    // [B, S]
  const float* Wh   = (const float*)d_in[3];  // [D, E]
  const float* Ws   = (const float*)d_in[4];  // [D, D]
  const float* v    = (const float*)d_in[5];  // [D]

  float* out_ctx  = (float*)d_out;            // [B, E]
  float* out_attn = (float*)d_out + B_ * E_;  // [B, S] (energy staged, softmaxed in place)

  // workspace layout (bytes)
  char* ws = (char*)d_ws;
  u16* encb = (u16*)ws;                                   // 128 MB
  u16* whb  = (u16*)(ws + (size_t)B_ * S_ * E_ * 2);      //   2 MB
  u16* wsb  = (u16*)((char*)whb + (size_t)D_ * E_ * 2);   //   2 MB
  u16* dhb  = (u16*)((char*)wsb + (size_t)D_ * D_ * 2);   //  64 KB
  float* dec = (float*)((char*)dhb + (size_t)B_ * D_ * 2);// 128 KB

  // one-time f32 -> bf16 conversions (memory-bound)
  {
    const int n8_enc = (B_ * S_ * E_) / 8;
    const int n8_wh  = (D_ * E_) / 8;
    const int n8_ws  = (D_ * D_) / 8;
    const int n8_dh  = (B_ * D_) / 8;
    f32_to_bf16_kernel<<<(n8_enc + 255) / 256, 256, 0, stream>>>(enc, (u32*)encb, n8_enc);
    f32_to_bf16_kernel<<<(n8_wh  + 255) / 256, 256, 0, stream>>>(Wh,  (u32*)whb,  n8_wh);
    f32_to_bf16_kernel<<<(n8_ws  + 255) / 256, 256, 0, stream>>>(Ws,  (u32*)wsb,  n8_ws);
    f32_to_bf16_kernel<<<(n8_dh  + 255) / 256, 256, 0, stream>>>(dh,  (u32*)dhb,  n8_dh);
  }

  dec_proj_kernel<<<128, 32, 0, stream>>>(dhb, wsb, dec);
  energy_kernel<<<(B_ * S_) / 64, 256, 0, stream>>>(encb, whb, dec, v, out_attn);
  softmax_kernel<<<B_, 256, 0, stream>>>(out_attn, mask);
  context_kernel<<<dim3(B_, E_ / 256), 256, 0, stream>>>(enc, out_attn, out_ctx);
}
